// Hessian3_16501264351427
// MI455X (gfx1250) — matrix operand order repaired and checked by reading, not compiled, not run on hardware
//
#include <hip/hip_runtime.h>
#include <hip/hip_bf16.h>

typedef __attribute__((ext_vector_type(8)))  float    v8f;
typedef __attribute__((ext_vector_type(16))) _Float16 v16h;

// ---------------- workspace layout (bytes) ----------------
#define OFF_FRAG   0          // 141 frags * 1024B = 144384
#define OFF_B1CAT  144384     // 96 * 4
#define OFF_B2CAT  144768     // 16 * 4
#define OFF_L0     147456     // 15360 * 4
#define OFF_L1     208896     // 15360 * 3 * 4
#define OFF_STAGE  393216     // 71680 halfs = 143360 B

// frag table: mix(4x8)=32 @0 | g1W1(6x8)=48 @32 | g1W2(4x8)=32 @80
//  | w1cat(1x6)=6 @112 | w2cat(3x1)=3 @118 | hw1(1x2)=2 @121 | hw2(1x1)=1 @124
//  | g2mix(2x1)=2 @125 | g2W1(3x4)=12 @127 | g2W2(2x1)=2 @139   -> 141 total
#define FRAG_MIX   0
#define FRAG_G1W1  32
#define FRAG_G1W2  80
#define FRAG_W1CAT 112
#define FRAG_W2CAT 118
#define FRAG_HW1   121
#define FRAG_HW2   124
#define FRAG_G2MIX 125
#define FRAG_G2W1  127
#define FRAG_G2W2  139

__device__ __forceinline__ float siluf(float x) { return x / (1.0f + __expf(-x)); }

__device__ __forceinline__ void lds_fence() {
    asm volatile("s_wait_dscnt 0" ::: "memory");
}

__device__ __forceinline__ unsigned h2u(_Float16 h) {
    union { _Float16 h; unsigned short u; } c; c.h = h; return (unsigned)c.u;
}

// B fragment: pre-swizzled, 2 x b128 per lane
__device__ __forceinline__ v16h load_frag(const unsigned* fragArea, int fragIdx, int lane) {
    union { v16h h; uint4 q[2]; } u;
    const uint4* p = (const uint4*)(fragArea + fragIdx * 256 + lane * 8);
    u.q[0] = p[0];
    u.q[1] = p[1];
    return u.h;
}

// A fragment (16x32 f16) from a global f32 row (this lane's row), converting.
// lane<16 holds K {0..7,16..23}+kbase ; lane>=16 holds K {8..15,24..31}+kbase
__device__ __forceinline__ v16h a_frag_global(const float* row, int kbase, int grp) {
    union { v16h h; _Float16 e[16]; } u;
    const float* p0 = row + kbase + grp * 8;
#pragma unroll
    for (int t = 0; t < 8; t++) u.e[t] = (_Float16)p0[t];
    const float* p1 = p0 + 16;
#pragma unroll
    for (int t = 0; t < 8; t++) u.e[8 + t] = (_Float16)p1[t];
    return u.h;
}

// A fragment from row-major f16 LDS buffer (ldHalfs multiple of 8, 16B aligned)
__device__ __forceinline__ v16h a_frag_lds(const _Float16* base, int ldHalfs, int kbase, int r, int grp) {
    union { v16h h; uint4 q[2]; } u;
    const _Float16* p = base + r * ldHalfs + kbase + grp * 8;
    u.q[0] = *(const uint4*)p;
    u.q[1] = *(const uint4*)(p + 16);
    return u.h;
}

#define WMMA(a, b, c) __builtin_amdgcn_wmma_f32_16x16x32_f16(false, (a), false, (b), (short)0, (c), false, false)

// ================= kernel 0a: build padded f16 weights + cat biases =================
__global__ void pack_fill(const float* g1mix, const float* g1W1, const float* g1W2,
                          const float* vvW1, const float* vrW1, const float* sW1,
                          const float* vvW2, const float* vrW2, const float* sW2,
                          const float* hW1, const float* hW2,
                          const float* vvb1, const float* vrb1, const float* sb1,
                          const float* vvb2, const float* vrb2, const float* sb2,
                          const float* g2mix, const float* g2W1, const float* g2W2,
                          _Float16* stage, float* b1cat, float* b2cat) {
    int tid = threadIdx.x;
    // mixP @0 : 128x128 direct
    for (int i = tid; i < 128 * 128; i += 256) stage[i] = (_Float16)g1mix[i];
    // w1P @16384 : 192x128 direct
    for (int i = tid; i < 192 * 128; i += 256) stage[16384 + i] = (_Float16)g1W1[i];
    // w2P @40960 : 128x128 direct
    for (int i = tid; i < 128 * 128; i += 256) stage[40960 + i] = (_Float16)g1W2[i];
    // w1catP @57344 : 32x96 block-diag of vv_W1(9x30), vr_W1(9x30), s_W1(2x30)
    for (int i = tid; i < 32 * 96; i += 256) {
        int k = i / 96, j = i % 96;
        float v = 0.f;
        if (j < 30)      { if (k < 9)             v = vvW1[k * 30 + j]; }
        else if (j < 60) { if (k >= 9 && k < 18)  v = vrW1[(k - 9) * 30 + (j - 30)]; }
        else if (j < 90) { if (k >= 18 && k < 20) v = sW1[(k - 18) * 30 + (j - 60)]; }
        stage[57344 + i] = (_Float16)v;
    }
    // w2catP @60416 : 96x16 stacked vv_W2(30x9), vr_W2, s_W2
    for (int i = tid; i < 96 * 16; i += 256) {
        int k = i / 16, j = i % 16;
        float v = 0.f;
        if (j < 9) {
            if (k < 30)      v = vvW2[k * 9 + j];
            else if (k < 60) v = vrW2[(k - 30) * 9 + j];
            else if (k < 90) v = sW2[(k - 60) * 9 + j];
        }
        stage[60416 + i] = (_Float16)v;
    }
    // hw1P @61952 : 32x32 from h_W1(9x30)
    for (int i = tid; i < 32 * 32; i += 256) {
        int k = i / 32, j = i % 32;
        stage[61952 + i] = (_Float16)((k < 9 && j < 30) ? hW1[k * 30 + j] : 0.f);
    }
    // hw2P @62976 : 32x16 from h_W2(30x9)
    for (int i = tid; i < 32 * 16; i += 256) {
        int k = i / 16, j = i % 16;
        stage[62976 + i] = (_Float16)((k < 30 && j < 9) ? hW2[k * 9 + j] : 0.f);
    }
    // g2mixP @63488 : 64x16 from g2_mix_W(64x2)
    for (int i = tid; i < 64 * 16; i += 256) {
        int k = i / 16, j = i % 16;
        stage[63488 + i] = (_Float16)((j < 2) ? g2mix[k * 2 + j] : 0.f);
    }
    // g2W1P @64512 : 96x64 from g2_s_W1(65x64)
    for (int i = tid; i < 96 * 64; i += 256) {
        int k = i / 64, j = i % 64;
        stage[64512 + i] = (_Float16)((k < 65) ? g2W1[k * 64 + j] : 0.f);
    }
    // g2W2P @70656 : 64x16 from g2_s_W2(64x2)
    for (int i = tid; i < 64 * 16; i += 256) {
        int k = i / 16, j = i % 16;
        stage[70656 + i] = (_Float16)((j < 2) ? g2W2[k * 2 + j] : 0.f);
    }
    // cat biases
    for (int j = tid; j < 96; j += 256) {
        float v = 0.f;
        if (j < 30) v = vvb1[j]; else if (j < 60) v = vrb1[j - 30]; else if (j < 90) v = sb1[j - 60];
        b1cat[j] = v;
    }
    for (int j = tid; j < 16; j += 256)
        b2cat[j] = (j < 9) ? (vvb2[j] + vrb2[j] + sb2[j]) : 0.f;
}

// ================= kernel 0b: swizzle padded f16 into WMMA B-fragment layout ========
__global__ void pack_swizzle(const _Float16* stage, unsigned* fragArea) {
    const int bases[11] = {0, 32, 80, 112, 118, 121, 124, 125, 127, 139, 141};
    const int NTs[10]   = {8, 8, 8, 6, 1, 2, 1, 1, 4, 1};
    const int offs[10]  = {0, 16384, 40960, 57344, 60416, 61952, 62976, 63488, 64512, 70656};
    int total = 141 * 256;
    for (int t = blockIdx.x * blockDim.x + threadIdx.x; t < total; t += gridDim.x * blockDim.x) {
        int f = t >> 8, within = t & 255, lane = within >> 3, v = within & 7;
        int m = 0;
        while (f >= bases[m + 1]) m++;
        int lf = f - bases[m];
        int NT = NTs[m];
        int kt = lf / NT, nt = lf % NT;
        int ld = NT * 16;
        int col = nt * 16 + (lane & 15);
        int krow = kt * 32 + ((v < 4) ? 0 : 8) + ((lane < 16) ? 0 : 8) + 2 * v;
        const _Float16* src = stage + offs[m];
        unsigned lo = h2u(src[krow * ld + col]);
        unsigned hi = h2u(src[(krow + 1) * ld + col]);
        fragArea[t] = lo | (hi << 16);
    }
}

// ================= kernel 1: per-atom gated blocks -> l0, l1 =================
#define K1_WAVES 2
__global__ __launch_bounds__(64) void k_atoms(
        const float* __restrict__ srep, const float* __restrict__ vrep,
        const float* __restrict__ g1b1, const float* __restrict__ g1b2,
        const float* __restrict__ g2b1, const float* __restrict__ g2b2,
        const unsigned* __restrict__ fragArea,
        float* __restrict__ l0ws, float* __restrict__ l1ws, int ntiles) {
    __shared__ __align__(16) _Float16 xbuf[K1_WAVES][16 * 192];
    __shared__ __align__(16) _Float16 hbuf[K1_WAVES][16 * 128];
    __shared__ __align__(16) _Float16 vwbuf[K1_WAVES][3 * 16 * 64];
    __shared__ __align__(16) _Float16 v1buf[K1_WAVES][3 * 16 * 64];
    __shared__ __align__(16) _Float16 x2buf[K1_WAVES][16 * 96];
    __shared__ __align__(16) _Float16 h2buf[K1_WAVES][16 * 64];
    __shared__ float w2v[K1_WAVES][3 * 16];

    int wave = threadIdx.x >> 5, lane = threadIdx.x & 31;
    int r = lane & 15, grp = lane >> 4;
    // clear x2buf once: padded K-columns (65..95) must not contain NaN garbage
    for (int i = lane; i < 16 * 96; i += 32) x2buf[wave][i] = (_Float16)0.f;
    lds_fence();
    float b2l0 = g2b2[0], b2l1 = g2b2[1];

    for (int tile = blockIdx.x * K1_WAVES + wave; tile < ntiles; tile += gridDim.x * K1_WAVES) {
        int a0 = tile * 16;
        // ---- stage 1: vmix = v[:,d,:] @ mixW ; accumulate vV^2, stash vW ----
        v8f vnacc[4];
#pragma unroll
        for (int q = 0; q < 4; q++) vnacc[q] = v8f{};
        v16h af[3][4];
#pragma unroll
        for (int d = 0; d < 3; d++)
#pragma unroll
            for (int kt = 0; kt < 4; kt++)
                af[d][kt] = a_frag_global(vrep + ((size_t)(a0 + r) * 3 + d) * 128, kt * 32, grp);
        for (int nt = 0; nt < 8; nt++) {
            v8f acc[3] = {v8f{}, v8f{}, v8f{}};
            for (int kt = 0; kt < 4; kt++) {
                v16h bf = load_frag(fragArea, FRAG_MIX + kt * 8 + nt, lane);
#pragma unroll
                for (int d = 0; d < 3; d++) acc[d] = WMMA(af[d][kt], bf, acc[d]);
            }
            if (nt < 4) {
#pragma unroll
                for (int d = 0; d < 3; d++) vnacc[nt] += acc[d] * acc[d];
            } else {
#pragma unroll
                for (int d = 0; d < 3; d++)
#pragma unroll
                    for (int e = 0; e < 8; e++)
                        vwbuf[wave][(d * 16 + e + grp * 8) * 64 + (nt - 4) * 16 + r] = (_Float16)acc[d][e];
            }
        }
        // ---- build x = [s | vn] (16x192 f16) ----
        for (int idx = lane; idx < 16 * 32; idx += 32) {
            int row = idx >> 5, c4 = (idx & 31) * 4;
            float4 val = *(const float4*)(srep + (size_t)(a0 + row) * 128 + c4);
            xbuf[wave][row * 192 + c4 + 0] = (_Float16)val.x;
            xbuf[wave][row * 192 + c4 + 1] = (_Float16)val.y;
            xbuf[wave][row * 192 + c4 + 2] = (_Float16)val.z;
            xbuf[wave][row * 192 + c4 + 3] = (_Float16)val.w;
        }
#pragma unroll
        for (int nt = 0; nt < 4; nt++)
#pragma unroll
            for (int e = 0; e < 8; e++)
                xbuf[wave][(e + grp * 8) * 192 + 128 + nt * 16 + r] = (_Float16)sqrtf(vnacc[nt][e]);
        lds_fence();
        // ---- GEMM1: x(16x192) @ g1_s_W1 -> silu -> hbuf ----
        v16h a1[6];
#pragma unroll
        for (int kt = 0; kt < 6; kt++) a1[kt] = a_frag_lds(xbuf[wave], 192, kt * 32, r, grp);
        for (int nt = 0; nt < 8; nt++) {
            v8f acc = v8f{};
            for (int kt = 0; kt < 6; kt++)
                acc = WMMA(a1[kt], load_frag(fragArea, FRAG_G1W1 + kt * 8 + nt, lane), acc);
            float b = g1b1[nt * 16 + r];
#pragma unroll
            for (int e = 0; e < 8; e++)
                hbuf[wave][(e + grp * 8) * 128 + nt * 16 + r] = (_Float16)siluf(acc[e] + b);
        }
        lds_fence();
        // ---- GEMM2: h(16x128) @ g1_s_W2 -> s1(silu)->x2buf ; gate*vW -> v1buf ----
        v16h a2[4];
#pragma unroll
        for (int kt = 0; kt < 4; kt++) a2[kt] = a_frag_lds(hbuf[wave], 128, kt * 32, r, grp);
        for (int nt = 0; nt < 8; nt++) {
            v8f acc = v8f{};
            for (int kt = 0; kt < 4; kt++)
                acc = WMMA(a2[kt], load_frag(fragArea, FRAG_G1W2 + kt * 8 + nt, lane), acc);
            float b = g1b2[nt * 16 + r];
#pragma unroll
            for (int e = 0; e < 8; e++) {
                float x = acc[e] + b;
                int M = e + grp * 8;
                if (nt < 4) {
                    x2buf[wave][M * 96 + nt * 16 + r] = (_Float16)siluf(x);
                } else {
                    int k = (nt - 4) * 16 + r;
#pragma unroll
                    for (int d = 0; d < 3; d++)
                        v1buf[wave][(d * 16 + M) * 64 + k] =
                            (_Float16)(x * (float)vwbuf[wave][(d * 16 + M) * 64 + k]);
                }
            }
        }
        lds_fence();
        // ---- stage 2 mix: v1[:,d,:](16x64) @ g2mixP(64x16) : col0=vV2, col1=vW2 ----
        v8f am[3];
#pragma unroll
        for (int d = 0; d < 3; d++) {
            v8f acc = v8f{};
#pragma unroll
            for (int kt = 0; kt < 2; kt++)
                acc = WMMA(a_frag_lds(&v1buf[wave][d * 16 * 64], 64, kt * 32, r, grp),
                           load_frag(fragArea, FRAG_G2MIX + kt, lane), acc);
            am[d] = acc;
        }
        if (r == 0) {
#pragma unroll
            for (int e = 0; e < 8; e++)
                x2buf[wave][(e + grp * 8) * 96 + 64] =
                    (_Float16)sqrtf(am[0][e] * am[0][e] + am[1][e] * am[1][e] + am[2][e] * am[2][e]);
        }
        if (r == 1) {
#pragma unroll
            for (int d = 0; d < 3; d++)
#pragma unroll
                for (int e = 0; e < 8; e++)
                    w2v[wave][d * 16 + e + grp * 8] = am[d][e];
        }
        lds_fence();
        // ---- stage 2 MLP: x2(16x96) @ g2W1P(96x64), silu -> h2buf ----
        v16h ax[3];
#pragma unroll
        for (int kt = 0; kt < 3; kt++) ax[kt] = a_frag_lds(x2buf[wave], 96, kt * 32, r, grp);
#pragma unroll
        for (int nt = 0; nt < 4; nt++) {
            v8f acc = v8f{};
#pragma unroll
            for (int kt = 0; kt < 3; kt++)
                acc = WMMA(ax[kt], load_frag(fragArea, FRAG_G2W1 + kt * 4 + nt, lane), acc);
            float b = g2b1[nt * 16 + r];
#pragma unroll
            for (int e = 0; e < 8; e++)
                h2buf[wave][(e + grp * 8) * 64 + nt * 16 + r] = (_Float16)siluf(acc[e] + b);
        }
        lds_fence();
        // ---- stage 2 out: h2(16x64) @ g2W2P(64x16) : col0=l0, col1=gate2 ----
        v8f accy = v8f{};
#pragma unroll
        for (int kt = 0; kt < 2; kt++)
            accy = WMMA(a_frag_lds(h2buf[wave], 64, kt * 32, r, grp),
                        load_frag(fragArea, FRAG_G2W2 + kt, lane), accy);
        if (r == 0) {
#pragma unroll
            for (int e = 0; e < 8; e++)
                l0ws[a0 + e + grp * 8] = accy[e] + b2l0;
        }
        if (r == 1) {
#pragma unroll
            for (int e = 0; e < 8; e++) {
                int M = e + grp * 8;
                float y1 = accy[e] + b2l1;
#pragma unroll
                for (int d = 0; d < 3; d++)
                    l1ws[(size_t)(a0 + M) * 3 + d] = y1 * w2v[wave][d * 16 + M];
            }
        }
    }
}

// ================= kernel 2: pairwise Hessian blocks =================
#define K2_WAVES 8
__global__ __launch_bounds__(256) void k_pairs(
        const float* __restrict__ pos, const float* __restrict__ l0ws, const float* __restrict__ l1ws,
        const unsigned* __restrict__ fragArea, const float* __restrict__ b1cat,
        const float* __restrict__ b2cat, const float* __restrict__ hb1, const float* __restrict__ hb2,
        float* __restrict__ out, int ntiles) {
    __shared__ __align__(16) _Float16 st1[K2_WAVES][16 * 96];
    __shared__ __align__(16) _Float16 stm[K2_WAVES][16 * 32];
    __shared__ __align__(16) _Float16 sth[K2_WAVES][16 * 32];
    int wave = threadIdx.x >> 5, lane = threadIdx.x & 31;
    int r = lane & 15, grp = lane >> 4;
    for (int i = threadIdx.x; i < K2_WAVES * 16 * 32; i += blockDim.x)
        (&stm[0][0])[i] = (_Float16)0.f;
    __syncthreads();

    v16h w1f[6], w2f[3], h1f[2], h2f;
#pragma unroll
    for (int nt = 0; nt < 6; nt++) w1f[nt] = load_frag(fragArea, FRAG_W1CAT + nt, lane);
#pragma unroll
    for (int kt = 0; kt < 3; kt++) w2f[kt] = load_frag(fragArea, FRAG_W2CAT + kt, lane);
    h1f[0] = load_frag(fragArea, FRAG_HW1 + 0, lane);
    h1f[1] = load_frag(fragArea, FRAG_HW1 + 1, lane);
    h2f = load_frag(fragArea, FRAG_HW2, lane);
    float bb2 = (r < 9) ? b2cat[r] : 0.f;
    float hbb2 = (r < 9) ? hb2[r] : 0.f;
    float b1c[6];
#pragma unroll
    for (int nt = 0; nt < 6; nt++) b1c[nt] = b1cat[nt * 16 + r];
    float hb1c[2];
    hb1c[0] = hb1[r];
    hb1c[1] = (16 + r < 30) ? hb1[16 + r] : 0.f;

    for (int t = blockIdx.x * K2_WAVES + wave; t < ntiles; t += gridDim.x * K2_WAVES) {
        int b = t / 400, rem = t % 400, i = rem / 5, j0 = (rem % 5) * 16;
        int ai = b * 80 + i, aj = b * 80 + j0 + r;
        float li[3], lj[3], pj[3];
#pragma unroll
        for (int d = 0; d < 3; d++) {
            li[d] = l1ws[(size_t)ai * 3 + d];
            lj[d] = l1ws[(size_t)aj * 3 + d];
            pj[d] = pos[(size_t)aj * 3 + d];
        }
        float l0i = l0ws[ai], l0j = l0ws[aj];
        float f[20];
#pragma unroll
        for (int a = 0; a < 3; a++)
#pragma unroll
            for (int c = 0; c < 3; c++) {
                f[a * 3 + c] = li[a] * lj[c];
                f[9 + a * 3 + c] = li[a] * pj[c];
            }
        f[18] = l0i;
        f[19] = l0j;
        union { v16h h; _Float16 e[16]; } ua;
#pragma unroll
        for (int v = 0; v < 8; v++)
#pragma unroll
            for (int hh = 0; hh < 2; hh++) {
                int k = (v < 4) ? ((grp ? 8 : 0) + 2 * v + hh) : ((grp ? 24 : 16) + 2 * (v - 4) + hh);
                ua.e[v * 2 + hh] = (_Float16)((k < 20) ? f[k] : 0.f);
            }
        // GEMM1: features(16x32) @ W1cat(32x96), silu -> st1
#pragma unroll
        for (int nt = 0; nt < 6; nt++) {
            v8f acc = v8f{};
            acc = WMMA(ua.h, w1f[nt], acc);
#pragma unroll
            for (int e = 0; e < 8; e++)
                st1[wave][(e + grp * 8) * 96 + nt * 16 + r] = (_Float16)siluf(acc[e] + b1c[nt]);
        }
        lds_fence();
        // GEMM2: st1(16x96) @ W2cat(96x16) -> mini
        v8f acc2 = v8f{};
#pragma unroll
        for (int kt = 0; kt < 3; kt++)
            acc2 = WMMA(a_frag_lds(st1[wave], 96, kt * 32, r, grp), w2f[kt], acc2);
        if (r < 9) {
#pragma unroll
            for (int e = 0; e < 8; e++)
                stm[wave][(e + grp * 8) * 32 + r] = (_Float16)(acc2[e] + bb2);
        }
        lds_fence();
        // GEMM3: mini(16x32) @ hW1(32x32), silu -> sth
        {
            v16h a3 = a_frag_lds(stm[wave], 32, 0, r, grp);
#pragma unroll
            for (int nt = 0; nt < 2; nt++) {
                v8f acc = v8f{};
                acc = WMMA(a3, h1f[nt], acc);
#pragma unroll
                for (int e = 0; e < 8; e++)
                    sth[wave][(e + grp * 8) * 32 + nt * 16 + r] = (_Float16)siluf(acc[e] + hb1c[nt]);
            }
        }
        lds_fence();
        // GEMM4: sth(16x32) @ hW2(32x16) -> out (transposed Hessian scatter)
        {
            v16h a4 = a_frag_lds(sth[wave], 32, 0, r, grp);
            v8f o = v8f{};
            o = WMMA(a4, h2f, o);
            if (r < 9) {
                int k = r / 3, l = r % 3;
                size_t rowbase = ((size_t)(b * 80 + i) * 3 + k) * 240;
#pragma unroll
                for (int e = 0; e < 8; e++) {
                    int jj = j0 + e + grp * 8;
                    out[rowbase + (size_t)jj * 3 + l] = o[e] + hbb2;
                }
            }
        }
    }
}

extern "C" void kernel_launch(void* const* d_in, const int* in_sizes, int n_in,
                              void* d_out, int out_size, void* d_ws, size_t ws_size,
                              hipStream_t stream) {
    const float* pos  = (const float*)d_in[0];
    const float* srep = (const float*)d_in[1];
    const float* vrep = (const float*)d_in[2];
    const float* g1mix = (const float*)d_in[4];
    const float* g1W1 = (const float*)d_in[5];
    const float* g1b1 = (const float*)d_in[6];
    const float* g1W2 = (const float*)d_in[7];
    const float* g1b2 = (const float*)d_in[8];
    const float* g2mix = (const float*)d_in[9];
    const float* g2W1 = (const float*)d_in[10];
    const float* g2b1 = (const float*)d_in[11];
    const float* g2W2 = (const float*)d_in[12];
    const float* g2b2 = (const float*)d_in[13];
    const float* vvW1 = (const float*)d_in[14]; const float* vvb1 = (const float*)d_in[15];
    const float* vvW2 = (const float*)d_in[16]; const float* vvb2 = (const float*)d_in[17];
    const float* vrW1 = (const float*)d_in[18]; const float* vrb1 = (const float*)d_in[19];
    const float* vrW2 = (const float*)d_in[20]; const float* vrb2 = (const float*)d_in[21];
    const float* sW1  = (const float*)d_in[22]; const float* sb1  = (const float*)d_in[23];
    const float* sW2  = (const float*)d_in[24]; const float* sb2  = (const float*)d_in[25];
    const float* hW1  = (const float*)d_in[26]; const float* hb1  = (const float*)d_in[27];
    const float* hW2  = (const float*)d_in[28]; const float* hb2  = (const float*)d_in[29];

    char* ws = (char*)d_ws;
    _Float16* stage = (_Float16*)(ws + OFF_STAGE);
    unsigned* frags = (unsigned*)(ws + OFF_FRAG);
    float* b1cat = (float*)(ws + OFF_B1CAT);
    float* b2cat = (float*)(ws + OFF_B2CAT);
    float* l0ws = (float*)(ws + OFF_L0);
    float* l1ws = (float*)(ws + OFF_L1);

    pack_fill<<<1, 256, 0, stream>>>(g1mix, g1W1, g1W2, vvW1, vrW1, sW1, vvW2, vrW2, sW2,
                                     hW1, hW2, vvb1, vrb1, sb1, vvb2, vrb2, sb2,
                                     g2mix, g2W1, g2W2, stage, b1cat, b2cat);
    pack_swizzle<<<32, 256, 0, stream>>>(stage, frags);
    k_atoms<<<480, 64, 0, stream>>>(srep, vrep, g1b1, g1b2, g2b1, g2b2,
                                    frags, l0ws, l1ws, 960);
    k_pairs<<<960, 256, 0, stream>>>(pos, l0ws, l1ws, frags, b1cat, b2cat, hb1, hb2,
                                     (float*)d_out, 76800);
}
